// ContextualLoss_21732534518053
// MI455X (gfx1250) — matrix operand order repaired and checked by reading, not compiled, not run on hardware
//
#include <hip/hip_runtime.h>
#include <hip/hip_bf16.h>
#include <hip/hip_fp16.h>

// ---------------------------------------------------------------------------
// ContextualLoss, fused for MI455X (gfx1250, wave32, WMMA).
//   X,T : (40, 16384) fp32 -> column-normalized f16 (K padded 40->64)
//   S = Xn^T Tn  : 16384x16384, K=40 GEMM, never materialized.
//   Two register-resident passes per 16-row strip:
//     pass1: row minima of S;  pass2: sum/max of exp((1 - S/(min+eps))/h)
//   loss = -log( max_{i,j} w_ij / rowsum_i )
// ---------------------------------------------------------------------------

#define D_ROWS   40
#define NCOLS    16384
#define KPAD     64
#define EPSILON  1e-5f
#define COS_EPS  1e-8f
#define INV_H    5.0f            // 1 / h, h = 0.2

typedef __attribute__((ext_vector_type(16))) _Float16 v16h;
typedef __attribute__((ext_vector_type(8)))  float    v8f;

union Frag { unsigned int u[8]; v16h h; };

// ---------------------------------------------------------------- init ------
__global__ void ctx_init(float* gmax) { *gmax = 0.0f; }

// ------------------------------------------------------------- prepare ------
// One thread per column j: mu (from target), subtract, L2-normalize columns,
// write f16 rows of length KPAD (j-major so WMMA fragment dwords are k-pairs).
__global__ __launch_bounds__(256)
void ctx_prepare(const float* __restrict__ input,
                 const float* __restrict__ target,
                 _Float16* __restrict__ Xh,
                 _Float16* __restrict__ Th) {
    const int j = blockIdx.x * blockDim.x + threadIdx.x;   // 0..16383 exact

    float tv[D_ROWS], xv[D_ROWS];
    float mu = 0.0f;
#pragma unroll
    for (int i = 0; i < D_ROWS; ++i) {
        tv[i] = target[(size_t)(10 + i) * NCOLS + j];
        xv[i] = input [(size_t)(10 + i) * NCOLS + j];
        mu += tv[i];
    }
    mu *= (1.0f / (float)D_ROWS);

    float nt = 0.0f, nx = 0.0f;
#pragma unroll
    for (int i = 0; i < D_ROWS; ++i) {
        float t = tv[i] - mu;
        float x = xv[i] - mu;
        tv[i] = t; xv[i] = x;
        nt += t * t; nx += x * x;
    }
    const float rt = 1.0f / fmaxf(sqrtf(nt), COS_EPS);
    const float rx = 1.0f / fmaxf(sqrtf(nx), COS_EPS);

    _Float16* xrow = Xh + (size_t)j * KPAD;
    _Float16* trow = Th + (size_t)j * KPAD;
#pragma unroll
    for (int i = 0; i < D_ROWS; ++i) {
        xrow[i] = (_Float16)(xv[i] * rx);
        trow[i] = (_Float16)(tv[i] * rt);
    }
#pragma unroll
    for (int i = D_ROWS; i < KPAD; ++i) {
        xrow[i] = (_Float16)0.0f;
        trow[i] = (_Float16)0.0f;
    }
}

// ---------------------------------------------------------------- main ------
// 128 blocks x 256 threads (8 waves). Wave w owns rows [blk*128+w*16, +16).
// Per 16-col N tile: two v_wmma_f32_16x16x32_f16 cover K=64.
__global__ __launch_bounds__(256)
void ctx_main(const _Float16* __restrict__ Xh,
              const _Float16* __restrict__ Th,
              float* __restrict__ gmax) {
    const int lane = threadIdx.x & 31;
    const int wave = threadIdx.x >> 5;        // 0..7
    const int half = lane >> 4;               // 0/1  (lane half-group)
    const int l16  = lane & 15;

    const int mBase = blockIdx.x * 128 + wave * 16;

    const unsigned int* Xu = (const unsigned int*)Xh;   // dword: [j*32 + k/2]
    const unsigned int* Tu = (const unsigned int*)Th;

    // --- A fragments (16x32 f16 layout): lane<16 -> M=lane, K pairs
    //     VGPR i: k = (i<4 ? 2i : 2i+8) + half*8
    Frag a0, a1;
    {
        const size_t mrow = (size_t)(mBase + l16) * 32;
#pragma unroll
        for (int i = 0; i < 8; ++i) {
            const int k = ((i < 4) ? 2 * i : 2 * i + 8) + half * 8;
            a0.u[i] = Xu[mrow + (k >> 1)];
            a1.u[i] = Xu[mrow + ((k + 32) >> 1)];
        }
    }

    const int NT = NCOLS / 16;                 // 1024 N tiles

    // ============================ pass 1: row minima =======================
    float rowMin[8];
#pragma unroll
    for (int r = 0; r < 8; ++r) rowMin[r] = 3.4e38f;

    for (int nt = 0; nt < NT; ++nt) {
        const size_t nrow = (size_t)(nt * 16 + l16) * 32;
        if (nt + 1 < NT)
            __builtin_prefetch(Tu + nrow + 16 * 32, 0, 0);  // next B tile -> L2/L0

        Frag b0, b1;                           // B 32x16: VGPR i -> K=(i<4?2i:2i+8)+half*8
#pragma unroll
        for (int i = 0; i < 8; ++i) {
            const int k = ((i < 4) ? 2 * i : 2 * i + 8) + half * 8;
            b0.u[i] = Tu[nrow + (k >> 1)];
            b1.u[i] = Tu[nrow + ((k + 32) >> 1)];
        }
        v8f c = {};
        c = __builtin_amdgcn_wmma_f32_16x16x32_f16(false, a0.h, false, b0.h,
                                                   (short)0, c, false, false);
        c = __builtin_amdgcn_wmma_f32_16x16x32_f16(false, a1.h, false, b1.h,
                                                   (short)0, c, false, false);
#pragma unroll
        for (int r = 0; r < 8; ++r) rowMin[r] = fminf(rowMin[r], c[r]);
    }
    // butterfly min across the 16-lane group -> every lane holds row min
#pragma unroll
    for (int m = 1; m <= 8; m <<= 1) {
#pragma unroll
        for (int r = 0; r < 8; ++r)
            rowMin[r] = fminf(rowMin[r], __shfl_xor(rowMin[r], m, 32));
    }
    float scale[8];                            // INV_H / (dmin + eps)
#pragma unroll
    for (int r = 0; r < 8; ++r) scale[r] = INV_H / (rowMin[r] + EPSILON);

    // ===================== pass 2: sum & max of w = exp(...) ===============
    float sumW[8], maxW[8];
#pragma unroll
    for (int r = 0; r < 8; ++r) { sumW[r] = 0.0f; maxW[r] = 0.0f; }

    for (int nt = 0; nt < NT; ++nt) {
        const size_t nrow = (size_t)(nt * 16 + l16) * 32;
        Frag b0, b1;
#pragma unroll
        for (int i = 0; i < 8; ++i) {
            const int k = ((i < 4) ? 2 * i : 2 * i + 8) + half * 8;
            b0.u[i] = Tu[nrow + (k >> 1)];
            b1.u[i] = Tu[nrow + ((k + 32) >> 1)];
        }
        v8f c = {};
        c = __builtin_amdgcn_wmma_f32_16x16x32_f16(false, a0.h, false, b0.h,
                                                   (short)0, c, false, false);
        c = __builtin_amdgcn_wmma_f32_16x16x32_f16(false, a1.h, false, b1.h,
                                                   (short)0, c, false, false);
#pragma unroll
        for (int r = 0; r < 8; ++r) {
            // w = exp((1 - S/(dmin+eps)) / h) = exp(INV_H - S*scale)
            const float w = __expf(INV_H - c[r] * scale[r]);
            sumW[r] += w;
            maxW[r]  = fmaxf(maxW[r], w);
        }
    }
#pragma unroll
    for (int m = 1; m <= 8; m <<= 1) {
#pragma unroll
        for (int r = 0; r < 8; ++r) {
            sumW[r] += __shfl_xor(sumW[r], m, 32);
            maxW[r]  = fmaxf(maxW[r], __shfl_xor(maxW[r], m, 32));
        }
    }

    // cx = maxW/sumW per row; global max via positive-float int atomicMax
    float localMax = 0.0f;
#pragma unroll
    for (int r = 0; r < 8; ++r) localMax = fmaxf(localMax, maxW[r] / sumW[r]);
#pragma unroll
    for (int m = 1; m <= 16; m <<= 1)
        localMax = fmaxf(localMax, __shfl_xor(localMax, m, 32));
    if (lane == 0)
        atomicMax((int*)gmax, __float_as_int(localMax));
}

// --------------------------------------------------------------- final ------
__global__ void ctx_final(const float* __restrict__ gmax,
                          float* __restrict__ out) {
    out[0] = -logf(*gmax);
}

// ---------------------------------------------------------------------------
extern "C" void kernel_launch(void* const* d_in, const int* in_sizes, int n_in,
                              void* d_out, int out_size, void* d_ws, size_t ws_size,
                              hipStream_t stream) {
    const float* input  = (const float*)d_in[0];
    const float* target = (const float*)d_in[1];
    float* out = (float*)d_out;

    // workspace: Xh (2MB) | Th (2MB) | gmax (4B)
    _Float16* Xh = (_Float16*)d_ws;
    _Float16* Th = Xh + (size_t)NCOLS * KPAD;
    float*  gmax = (float*)(Th + (size_t)NCOLS * KPAD);

    ctx_init<<<1, 1, 0, stream>>>(gmax);
    ctx_prepare<<<NCOLS / 256, 256, 0, stream>>>(input, target, Xh, Th);
    ctx_main<<<NCOLS / 128, 256, 0, stream>>>(Xh, Th, gmax);
    ctx_final<<<1, 1, 0, stream>>>(gmax, out);
}
